// WindowAttentionGlobal3D_75909251990230
// MI455X (gfx1250) — compile-verified, matrix operand order checked by reading, not run on hardware
//
#include <hip/hip_runtime.h>
#include <hip/hip_bf16.h>

typedef __attribute__((ext_vector_type(16))) _Float16 v16h;
typedef __attribute__((ext_vector_type(8)))  _Float16 v8h;
typedef __attribute__((ext_vector_type(8)))  float    v8f;
typedef __attribute__((ext_vector_type(4)))  int      v4i;

#define B_WIN   256     // B * B_dim
#define NTOK    196
#define CDIM    512
#define NHEAD   16
#define HDIM    32
#define MROWS   (B_WIN * NTOK)   // 50176
#define NPAD    208     // 13 * 16  (row/col tile pad of 196)
#define KPAD    224     // 7 * 32   (K pad of 196 for P@V)

#define LDS_FENCE() asm volatile("s_wait_dscnt 0" ::: "memory")

#if __has_builtin(__builtin_amdgcn_global_load_async_to_lds_b128) && \
    __has_builtin(__builtin_amdgcn_s_wait_asynccnt)
#define HAVE_ASYNC_LDS 1
#else
#define HAVE_ASYNC_LDS 0
#endif

#define AS1 __attribute__((address_space(1)))
#define AS3 __attribute__((address_space(3)))

#if HAVE_ASYNC_LDS
// 16-byte async DMA global -> LDS (per-lane addresses), tracked by ASYNCcnt
static __device__ __forceinline__ void async_cp16(const _Float16* g, _Float16* l) {
    __builtin_amdgcn_global_load_async_to_lds_b128((AS1 v4i*)g, (AS3 v4i*)l, 0, 0);
}
#endif

static __device__ __forceinline__ v16h frag2(const _Float16* p0, const _Float16* p1) {
    v8h a = *(const v8h*)p0;
    v8h b = *(const v8h*)p1;
    return __builtin_shufflevector(a, b, 0,1,2,3,4,5,6,7,8,9,10,11,12,13,14,15);
}

static __device__ __forceinline__ v8f wmma16(v16h a, v16h b, v8f c) {
    return __builtin_amdgcn_wmma_f32_16x16x32_f16(false, a, false, b, (short)0, c, false, false);
}

// relative position index for WINDOW_SIZE (4,7,7): dims (2h-1)=(2w-1)=13, table 7*169=1183
static __device__ __forceinline__ int rpi_index(int m, int c) {
    int t1 = m / 49, r1 = m - t1 * 49, h1 = r1 / 7, w1 = r1 - h1 * 7;
    int t2 = c / 49, r2 = c - t2 * 49, h2 = r2 / 7, w2 = r2 - h2 * 7;
    return (t1 - t2 + 3) * 169 + (h1 - h2 + 6) * 13 + (w1 - w2 + 6);
}

// ---------------------------------------------------------------- convert
__global__ void cvt_f16_kernel(const float* __restrict__ in, _Float16* __restrict__ out,
                               int n, float scale) {
    int i = blockIdx.x * blockDim.x + threadIdx.x;
    if (i < n) out[i] = (_Float16)(in[i] * scale);
}

// ---------------------------------------------------------------- kv GEMM
// A: (50176, 512) f16 rows of x, W: (1024, 512) f16, scatter to K/V (B_,H,N,hd) f16
__global__ void __launch_bounds__(256) gemm_kv_kernel(
        const _Float16* __restrict__ A, const _Float16* __restrict__ W,
        const float* __restrict__ bias,
        _Float16* __restrict__ Kh, _Float16* __restrict__ Vh) {
    __shared__ _Float16 sA[2][64 * 32];
    __shared__ _Float16 sB[2][64 * 32];
    const int mbase = blockIdx.x * 64;
    const int nbase = blockIdx.y * 64;
    const int tid = threadIdx.x;
    const int wave = tid >> 5, lane = tid & 31;
    const int mi = wave & 3, ni = wave >> 2;
    const int lm = lane & 15, hi = lane >> 4;
    const int cr = tid >> 2, cs = (tid & 3) * 8;
    const _Float16* gA = &A[(size_t)(mbase + cr) * CDIM + cs];
    const _Float16* gB = &W[(size_t)(nbase + cr) * CDIM + cs];
    v8f acc0 = {}; v8f acc1 = {};
#if HAVE_ASYNC_LDS
    async_cp16(gA, &sA[0][cr * 32 + cs]);
    async_cp16(gB, &sB[0][cr * 32 + cs]);
#endif
    for (int kk = 0; kk < CDIM / 32; ++kk) {
        const int cur = kk & 1;
#if HAVE_ASYNC_LDS
        if (kk + 1 < CDIM / 32) {
            async_cp16(gA + (kk + 1) * 32, &sA[cur ^ 1][cr * 32 + cs]);
            async_cp16(gB + (kk + 1) * 32, &sB[cur ^ 1][cr * 32 + cs]);
            __builtin_amdgcn_s_wait_asynccnt(2);   // chunk kk landed
        } else {
            __builtin_amdgcn_s_wait_asynccnt(0);
        }
#else
        *(v8h*)(&sA[cur][cr * 32 + cs]) = *(const v8h*)(gA + kk * 32);
        *(v8h*)(&sB[cur][cr * 32 + cs]) = *(const v8h*)(gB + kk * 32);
#endif
        __syncthreads();
        const _Float16* ar = &sA[cur][(mi * 16 + lm) * 32];
        v16h af = frag2(ar + hi * 8, ar + 16 + hi * 8);
        const _Float16* b0 = &sB[cur][(ni * 32 + lm) * 32];
        v16h bf0 = frag2(b0 + hi * 8, b0 + 16 + hi * 8);
        const _Float16* b1 = &sB[cur][(ni * 32 + 16 + lm) * 32];
        v16h bf1 = frag2(b1 + hi * 8, b1 + 16 + hi * 8);
        acc0 = wmma16(af, bf0, acc0);
        acc1 = wmma16(af, bf1, acc1);
        __syncthreads();
    }
    for (int r = 0; r < 8; ++r) {
        int gm = mbase + mi * 16 + r + hi * 8;
        int b_ = gm / NTOK, n = gm - b_ * NTOK;
        #pragma unroll
        for (int t = 0; t < 2; ++t) {
            int gc = nbase + ni * 32 + t * 16 + lm;
            float v = ((t == 0) ? acc0[r] : acc1[r]) + bias[gc];
            int sel = gc >> 9, h = (gc >> 5) & 15, d = gc & 31;
            _Float16* dst = sel ? Vh : Kh;
            dst[(((size_t)b_ * NHEAD + h) * NTOK + n) * HDIM + d] = (_Float16)v;
        }
    }
}

// ---------------------------------------------------------------- attention
// Q: (B,H,196,32) f16 pre-scaled; K,V: (B_,H,196,32) f16; O: (B_,196,512) f16
__global__ void __launch_bounds__(128) attn_kernel(
        const _Float16* __restrict__ Q, const _Float16* __restrict__ Kh,
        const _Float16* __restrict__ Vh, const float* __restrict__ bias_table,
        _Float16* __restrict__ O) {
    extern __shared__ unsigned char smem[];
    _Float16* q_s = (_Float16*)smem;                  // 208*32 f16
    _Float16* k_s = q_s + NPAD * HDIM;                // 208*32 f16
    _Float16* vT  = k_s + NPAD * HDIM;                // 32*224 f16
    float*    S_b = (float*)(vT + HDIM * KPAD);       // 4 waves * 16*208 f32
    _Float16* P_b = (_Float16*)(S_b + 4 * 16 * NPAD); // 4 waves * 16*224 f16
    float*    inv_s = (float*)(P_b + 4 * 16 * KPAD);  // 4 waves * 16 f32

    const int b_ = blockIdx.x;
    const int h  = blockIdx.y;
    const int b  = b_ >> 6;   // b_ / 64 windows per batch
    const int tid = threadIdx.x, wave = tid >> 5, lane = tid & 31;
    const int lm = lane & 15, hi = lane >> 4;
    const _Float16* qg = Q  + (((size_t)b  * NHEAD + h) * NTOK) * HDIM;
    const _Float16* kg = Kh + (((size_t)b_ * NHEAD + h) * NTOK) * HDIM;
    const _Float16* vg = Vh + (((size_t)b_ * NHEAD + h) * NTOK) * HDIM;

    // stage q, k (row-major, zero pad rows) and v transposed (zero pad cols)
    for (int idx = tid; idx < NPAD * 4; idx += 128) {
        int r = idx >> 2, seg = (idx & 3) * 8;
        if (r < NTOK) {
#if HAVE_ASYNC_LDS
            async_cp16(&qg[r * HDIM + seg], &q_s[r * HDIM + seg]);
            async_cp16(&kg[r * HDIM + seg], &k_s[r * HDIM + seg]);
#else
            *(v8h*)(&q_s[r * HDIM + seg]) = *(const v8h*)(&qg[r * HDIM + seg]);
            *(v8h*)(&k_s[r * HDIM + seg]) = *(const v8h*)(&kg[r * HDIM + seg]);
#endif
        } else {
            v8h z = {};
            *(v8h*)(&q_s[r * HDIM + seg]) = z;
            *(v8h*)(&k_s[r * HDIM + seg]) = z;
        }
    }
    for (int idx = tid; idx < HDIM * KPAD; idx += 128) {
        int d = idx / KPAD, n = idx - d * KPAD;
        vT[d * KPAD + n] = (n < NTOK) ? vg[n * HDIM + d] : (_Float16)0.f;
    }
#if HAVE_ASYNC_LDS
    __builtin_amdgcn_s_wait_asynccnt(0);
#endif
    __syncthreads();

    float*    Sw = S_b + wave * 16 * NPAD;
    _Float16* Pw = P_b + wave * 16 * KPAD;

    for (int t = wave; t < 13; t += 4) {
        // ---- S = q . k^T  (K = 32, one WMMA per 16x16 tile)
        const _Float16* ar = &q_s[(t * 16 + lm) * HDIM];
        v16h qf = frag2(ar + hi * 8, ar + 16 + hi * 8);
        for (int j = 0; j < 13; ++j) {
            const _Float16* br = &k_s[(j * 16 + lm) * HDIM];
            v16h kf = frag2(br + hi * 8, br + 16 + hi * 8);
            v8f acc = {};
            acc = wmma16(qf, kf, acc);
            #pragma unroll
            for (int r = 0; r < 8; ++r)
                Sw[(r + hi * 8) * NPAD + j * 16 + lm] = acc[r];
        }
        LDS_FENCE();

        // ---- softmax (2 lanes per row), bias gathered on the fly
        {
            int row = lane >> 1, halfsel = lane & 1;
            int m = t * 16 + row;
            bool valid = (m < NTOK);
            int c0 = halfsel ? 98 : 0;
            int c1 = halfsel ? NTOK : 98;
            float mx = -3.4e38f;
            if (valid)
                for (int c = c0; c < c1; ++c)
                    mx = fmaxf(mx, Sw[row * NPAD + c] +
                                   bias_table[rpi_index(m, c) * NHEAD + h]);
            mx = fmaxf(mx, __shfl_xor(mx, 1));
            float sum = 0.f;
            if (valid) {
                for (int c = c0; c < c1; ++c) {
                    float s = Sw[row * NPAD + c] +
                              bias_table[rpi_index(m, c) * NHEAD + h];
                    float p = __expf(s - mx);
                    Pw[row * KPAD + c] = (_Float16)p;
                    sum += p;
                }
            } else {
                for (int c = c0; c < c1; ++c) Pw[row * KPAD + c] = (_Float16)0.f;
            }
            if (halfsel)
                for (int c = NTOK; c < KPAD; ++c) Pw[row * KPAD + c] = (_Float16)0.f;
            sum += __shfl_xor(sum, 1);
            if (!halfsel) inv_s[wave * 16 + row] = valid ? (1.f / sum) : 0.f;
        }
        LDS_FENCE();

        // ---- out = P . v  (K = 224 in 7 chunks of 32; two 16-col tiles of hd)
        v8f acc0 = {}, acc1 = {};
        for (int kc = 0; kc < 7; ++kc) {
            const _Float16* pr = &Pw[lm * KPAD + kc * 32];
            v16h pf = frag2(pr + hi * 8, pr + 16 + hi * 8);
            const _Float16* v0 = &vT[lm * KPAD + kc * 32];
            v16h vf0 = frag2(v0 + hi * 8, v0 + 16 + hi * 8);
            const _Float16* v1 = &vT[(16 + lm) * KPAD + kc * 32];
            v16h vf1 = frag2(v1 + hi * 8, v1 + 16 + hi * 8);
            acc0 = wmma16(pf, vf0, acc0);
            acc1 = wmma16(pf, vf1, acc1);
        }
        #pragma unroll
        for (int r = 0; r < 8; ++r) {
            int m = t * 16 + r + hi * 8;
            if (m < NTOK) {
                float inv = inv_s[wave * 16 + r + hi * 8];
                size_t base = ((size_t)b_ * NTOK + m) * CDIM + h * HDIM;
                O[base + lm]      = (_Float16)(acc0[r] * inv);
                O[base + 16 + lm] = (_Float16)(acc1[r] * inv);
            }
        }
        LDS_FENCE();
    }
}

// ---------------------------------------------------------------- out projection
// A: (50176, 512) f16 attn-out, W: (512,512) f16, out f32 (50176,512)
__global__ void __launch_bounds__(256) gemm_proj_kernel(
        const _Float16* __restrict__ A, const _Float16* __restrict__ W,
        const float* __restrict__ bias, float* __restrict__ out) {
    __shared__ _Float16 sA[2][64 * 32];
    __shared__ _Float16 sB[2][64 * 32];
    const int mbase = blockIdx.x * 64;
    const int nbase = blockIdx.y * 64;
    const int tid = threadIdx.x;
    const int wave = tid >> 5, lane = tid & 31;
    const int mi = wave & 3, ni = wave >> 2;
    const int lm = lane & 15, hi = lane >> 4;
    const int cr = tid >> 2, cs = (tid & 3) * 8;
    const _Float16* gA = &A[(size_t)(mbase + cr) * CDIM + cs];
    const _Float16* gB = &W[(size_t)(nbase + cr) * CDIM + cs];
    v8f acc0 = {}; v8f acc1 = {};
#if HAVE_ASYNC_LDS
    async_cp16(gA, &sA[0][cr * 32 + cs]);
    async_cp16(gB, &sB[0][cr * 32 + cs]);
#endif
    for (int kk = 0; kk < CDIM / 32; ++kk) {
        const int cur = kk & 1;
#if HAVE_ASYNC_LDS
        if (kk + 1 < CDIM / 32) {
            async_cp16(gA + (kk + 1) * 32, &sA[cur ^ 1][cr * 32 + cs]);
            async_cp16(gB + (kk + 1) * 32, &sB[cur ^ 1][cr * 32 + cs]);
            __builtin_amdgcn_s_wait_asynccnt(2);
        } else {
            __builtin_amdgcn_s_wait_asynccnt(0);
        }
#else
        *(v8h*)(&sA[cur][cr * 32 + cs]) = *(const v8h*)(gA + kk * 32);
        *(v8h*)(&sB[cur][cr * 32 + cs]) = *(const v8h*)(gB + kk * 32);
#endif
        __syncthreads();
        const _Float16* ar = &sA[cur][(mi * 16 + lm) * 32];
        v16h af = frag2(ar + hi * 8, ar + 16 + hi * 8);
        const _Float16* b0 = &sB[cur][(ni * 32 + lm) * 32];
        v16h bf0 = frag2(b0 + hi * 8, b0 + 16 + hi * 8);
        const _Float16* b1 = &sB[cur][(ni * 32 + 16 + lm) * 32];
        v16h bf1 = frag2(b1 + hi * 8, b1 + 16 + hi * 8);
        acc0 = wmma16(af, bf0, acc0);
        acc1 = wmma16(af, bf1, acc1);
        __syncthreads();
    }
    for (int r = 0; r < 8; ++r) {
        int gm = mbase + mi * 16 + r + hi * 8;
        #pragma unroll
        for (int t = 0; t < 2; ++t) {
            int gc = nbase + ni * 32 + t * 16 + lm;
            out[(size_t)gm * CDIM + gc] = ((t == 0) ? acc0[r] : acc1[r]) + bias[gc];
        }
    }
}

extern "C" void kernel_launch(void* const* d_in, const int* in_sizes, int n_in,
                              void* d_out, int out_size, void* d_ws, size_t ws_size,
                              hipStream_t stream) {
    (void)in_sizes; (void)n_in; (void)out_size; (void)ws_size;
    const float* x       = (const float*)d_in[0];   // (256,196,512)
    const float* qglob   = (const float*)d_in[1];   // (4,1,16,196,32)
    const float* kv_w    = (const float*)d_in[2];   // (1024,512)
    const float* kv_b    = (const float*)d_in[3];   // (1024)
    const float* proj_w  = (const float*)d_in[4];   // (512,512)
    const float* proj_b  = (const float*)d_in[5];   // (512)
    const float* btab    = (const float*)d_in[6];   // (1183,16)
    float* out = (float*)d_out;

    char* ws = (char*)d_ws;
    const size_t BIG = (size_t)MROWS * CDIM * sizeof(_Float16);   // 51,380,224 B
    _Float16* xh     = (_Float16*)(ws);                 // x (f16); reused as attn-out
    _Float16* kh     = (_Float16*)(ws + BIG);           // (B_,H,N,hd)
    _Float16* vh     = (_Float16*)(ws + 2 * BIG);       // (B_,H,N,hd)
    _Float16* wkv_h  = (_Float16*)(ws + 3 * BIG);
    _Float16* wpr_h  = (_Float16*)(ws + 3 * BIG + 2 * 1024 * 512);
    _Float16* qh     = (_Float16*)(ws + 3 * BIG + 2 * 1024 * 512 + 2 * 512 * 512);

    const float qscale = 0.17677669529663687f;  // 1/sqrt(32)

    // 1. casts
    {
        int n = MROWS * CDIM;
        cvt_f16_kernel<<<(n + 255) / 256, 256, 0, stream>>>(x, xh, n, 1.f);
        n = 1024 * 512;
        cvt_f16_kernel<<<(n + 255) / 256, 256, 0, stream>>>(kv_w, wkv_h, n, 1.f);
        n = 512 * 512;
        cvt_f16_kernel<<<(n + 255) / 256, 256, 0, stream>>>(proj_w, wpr_h, n, 1.f);
        n = 4 * NHEAD * NTOK * HDIM;
        cvt_f16_kernel<<<(n + 255) / 256, 256, 0, stream>>>(qglob, qh, n, qscale);
    }
    // 2. kv projection
    gemm_kv_kernel<<<dim3(MROWS / 64, 1024 / 64), 256, 0, stream>>>(xh, wkv_h, kv_b, kh, vh);
    // 3. fused attention (writes attn-out over xh region)
    {
        const size_t smem = (size_t)NPAD * HDIM * 2 * 2      // q_s + k_s
                          + (size_t)HDIM * KPAD * 2          // vT
                          + 4u * 16 * NPAD * 4               // S per wave
                          + 4u * 16 * KPAD * 2               // P per wave
                          + 4u * 16 * 4;                     // inv per wave  = 123,136 B
        (void)hipFuncSetAttribute((const void*)attn_kernel,
                                  hipFuncAttributeMaxDynamicSharedMemorySize, (int)smem);
        attn_kernel<<<dim3(B_WIN, NHEAD), 128, smem, stream>>>(qh, kh, vh, btab, xh);
    }
    // 4. output projection -> f32 d_out
    gemm_proj_kernel<<<dim3(MROWS / 64, CDIM / 64), 256, 0, stream>>>(xh, wpr_h, proj_b, out);
}
